// custom_gru_901943132822
// MI455X (gfx1250) — matrix operand order repaired and checked by reading, not compiled, not run on hardware
//
#include <hip/hip_runtime.h>
#include <hip/hip_bf16.h>

// ---------------------------------------------------------------------------
// Autoregressive GRU for MI455X (gfx1250).
// bf16 WMMA (f32 accum) + TDM (tensor_load_to_lds) weight staging into LDS.
// Persistent kernel, grid-wide sync via monotonic atomic counter.
// ---------------------------------------------------------------------------

typedef __attribute__((ext_vector_type(16))) __bf16        v16bf;
typedef __attribute__((ext_vector_type(8)))  float         v8f;
typedef __attribute__((ext_vector_type(4)))  unsigned int  v4u;
typedef __attribute__((ext_vector_type(8)))  int           v8i;
typedef __attribute__((ext_vector_type(4)))  int           v4i;

#define D_    512          // feature dim (D == H)
#define B_    128          // batch
#define NBLK  24           // persistent workgroups == 24 (jt,nt) groups in phase A
#define TPB   128          // 4 waves per WG (wave32)
#define WSZ   262144       // 512*512 elements per weight matrix
#define PANEL 32768        // elements of one 64-col packed weight panel (64KB)

union FragU { v16bf v; v4u q[2]; };

__device__ __forceinline__ unsigned short f2bf(float f) {
  unsigned u = __float_as_uint(f);
  u += 0x7FFFu + ((u >> 16) & 1u);          // round-to-nearest-even
  return (unsigned short)(u >> 16);
}

// A-fragment (16x32 bf16) from ROW-MAJOR bf16 activations.
// lane 0-15: M=lane; K runs {half*8+0..7, 16+half*8+0..7} -> two b128 loads.
__device__ __forceinline__ v16bf load_a(const unsigned short* __restrict__ A,
                                        int row0, int kb, int lane) {
  const int m = row0 + (lane & 15);
  const unsigned short* p = A + m * D_ + (kb << 5) + ((lane >> 4) << 3);
  FragU f;
  f.q[0] = *(const v4u*)(p);
  f.q[1] = *(const v4u*)(p + 16);
  return f.v;
}

// B-fragment (32x16 bf16) from LDS-staged packed panel: lane holds N=lane%16,
// K = {(lane/16)*8+0..7, 16+(lane/16)*8+0..7} -> 32B contiguous per lane (ds_load_b128 x2).
__device__ __forceinline__ v16bf load_b_lds(const unsigned short* S,
                                            int sLocal, int kb, int lane) {
  const unsigned short* p = S + (((sLocal << 4) + kb) << 9) + (lane << 4);
  FragU f;
  f.q[0] = *(const v4u*)(p);
  f.q[1] = *(const v4u*)(p + 8);
  return f.v;
}

__device__ __forceinline__ v8f wmma_bf16(v16bf a, v16bf b, v8f c) {
  return __builtin_amdgcn_wmma_f32_16x16x32_bf16(
      false, a, false, b, (short)0, c, false, false);
}

// C += A[m0:m0+32, :] @ S (LDS panel, 512x64)
__device__ __forceinline__ void mm32x64_lds(const unsigned short* __restrict__ Abf,
                                            const unsigned short* S,
                                            int m0, int lane, v8f c[2][4]) {
  for (int kb = 0; kb < 16; ++kb) {
    v16bf a0 = load_a(Abf, m0,      kb, lane);
    v16bf a1 = load_a(Abf, m0 + 16, kb, lane);
#pragma unroll
    for (int s = 0; s < 4; ++s) {
      v16bf b = load_b_lds(S, s, kb, lane);
      c[0][s] = wmma_bf16(a0, b, c[0][s]);
      c[1][s] = wmma_bf16(a1, b, c[1][s]);
    }
  }
}

// C += A[m0:m0+16, :] @ S (LDS panel, 512x64)
__device__ __forceinline__ void mm16x64_lds(const unsigned short* __restrict__ Abf,
                                            const unsigned short* S,
                                            int m0, int lane, v8f c[4]) {
  for (int kb = 0; kb < 16; ++kb) {
    v16bf a = load_a(Abf, m0, kb, lane);
#pragma unroll
    for (int s = 0; s < 4; ++s) {
      v16bf b = load_b_lds(S, s, kb, lane);
      c[s] = wmma_bf16(a, b, c[s]);
    }
  }
}

// ---- Tensor Data Mover: 1-D contiguous copy global -> LDS (nbytes, 8B units).
// D# built per CDNA5 ISA §8: group0 {count=1, lds_addr, global_addr, type=2},
// group1 {data_size=8B, tensor_dim0=tile_dim0=n8, tensor_dim1=tile_dim1=1,
//         strides=n8}.  One issue per panel, tracked by TENSORcnt.
__device__ __forceinline__ void tdm_load(unsigned lds_off,
                                         const unsigned short* gsrc,
                                         unsigned nbytes) {
  unsigned long long ga = (unsigned long long)(uintptr_t)gsrc;
  unsigned n8 = nbytes >> 3;                         // 8-byte units
  v4u g0;
  g0.x = 1u;                                         // count=1, user mode
  g0.y = lds_off;                                    // LDS byte address
  g0.z = (unsigned)ga;                               // global_addr[31:0]
  g0.w = (unsigned)(ga >> 32) | 0x80000000u;         // global_addr[56:32] | type=2
  v8i g1;
  g1[0] = (int)0x00030000u;                          // data_size=3 (8B)
  g1[1] = (int)((n8 & 0xFFFFu) << 16);               // tensor_dim0[15:0] @ [31:16]
  g1[2] = (int)(((n8 >> 16) & 0xFFFFu) | (1u << 16));// tensor_dim0[31:16] | tensor_dim1=1
  g1[3] = (int)((n8 & 0xFFFFu) << 16);               // tile_dim0 @ [31:16]
  g1[4] = 1;                                         // tile_dim1=1, tile_dim2=0
  g1[5] = (int)n8;                                   // tensor_dim0_stride[31:0]
  g1[6] = (int)((n8 & 0xFFFFu) << 16);               // stride0[47:32]=0 | stride1[15:0]
  g1[7] = (int)(n8 >> 16);                           // stride1[47:16]
  v4i z4 = {0, 0, 0, 0};
#if __clang_major__ >= 23
  v8i z8 = {0, 0, 0, 0, 0, 0, 0, 0};
  __builtin_amdgcn_tensor_load_to_lds(g0, g1, z4, z4, z8, 0);
#else
  __builtin_amdgcn_tensor_load_to_lds(g0, g1, z4, z4, 0);
#endif
}

// ---------------- workspace layout ----------------
struct WS {
  unsigned short* wp[7];                  // packed bf16 weights: Wz Wr Wh Uz Ur Uh Wo
  unsigned short *xbf, *hbf, *rhbf;       // row-major bf16 activations [128,512]
  float *hbuf, *zbuf, *hpart;             // f32 state [128,512]
  unsigned* cnt;                          // grid barrier counter
};
__device__ __forceinline__ WS ws_layout(void* ws) {
  WS s;
  unsigned short* p = (unsigned short*)ws;
  for (int i = 0; i < 7; ++i) s.wp[i] = p + i * WSZ;
  s.xbf   = p + 7 * WSZ;
  s.hbf   = s.xbf + B_ * D_;
  s.rhbf  = s.hbf + B_ * D_;
  s.hbuf  = (float*)(s.rhbf + B_ * D_);
  s.zbuf  = s.hbuf + B_ * D_;
  s.hpart = s.zbuf + B_ * D_;
  s.cnt   = (unsigned*)(s.hpart + B_ * D_);
  return s;
}

__device__ __forceinline__ void grid_sync(unsigned* cnt, unsigned tgt) {
  __syncthreads();
  __threadfence();                               // release prior writes
  if (threadIdx.x == 0) {
    atomicAdd(cnt, 1u);
    while (__atomic_load_n(cnt, __ATOMIC_RELAXED) < tgt)
      __builtin_amdgcn_s_sleep(2);
  }
  __syncthreads();
  __threadfence();                               // acquire others' writes
}

// ---------------- setup kernels ----------------
__global__ void pack_weights(const float* __restrict__ Wz, const float* __restrict__ Wr,
                             const float* __restrict__ Wh, const float* __restrict__ Uz,
                             const float* __restrict__ Ur, const float* __restrict__ Uh,
                             const float* __restrict__ Wo, unsigned short* __restrict__ wpack) {
  const float* srcs[7] = {Wz, Wr, Wh, Uz, Ur, Uh, Wo};
  unsigned idx = blockIdx.x * 256u + threadIdx.x;     // 7*262144 total
  unsigned w = idx >> 18;
  if (w >= 7) return;
  unsigned d = idx & (WSZ - 1);
  unsigned panel = d >> 9;                            // (nb*16 + kb)
  unsigned le = d & 511;
  unsigned l = le >> 4, e = le & 15;
  unsigned nb = panel >> 4, kb = panel & 15;
  unsigned k = (kb << 5) + ((e >> 3) << 4) + ((l >> 4) << 3) + (e & 7);       // K row
  unsigned n = (nb << 4) + (l & 15);                  // N col
  wpack[w * WSZ + d] = f2bf(srcs[w][k * D_ + n]);
}

__global__ void init_state(const float* __restrict__ inputs, float* __restrict__ out,
                           const int* __restrict__ Tptr, void* ws) {
  WS s = ws_layout(ws);
  const int T = *Tptr;
  int idx = blockIdx.x * 256 + threadIdx.x;           // 0 .. 65535
  if (idx < B_ * D_) {
    int b = idx >> 9, dc = idx & 511;
    float x = inputs[idx];
    s.xbf[idx]  = f2bf(x);
    s.hbf[idx]  = 0;
    s.hbuf[idx] = 0.0f;
    out[(size_t)b * T * D_ + dc] = x;                 // t = 0 row
  }
  if (blockIdx.x == 0 && threadIdx.x == 0) *s.cnt = 0u;
}

// ---------------- persistent GRU kernel ----------------
__global__ __launch_bounds__(TPB) void gru_persistent(
    const float* __restrict__ bz, const float* __restrict__ br,
    const float* __restrict__ bh, const float* __restrict__ bo,
    float* __restrict__ out, const int* __restrict__ Tptr, void* ws) {
  WS s = ws_layout(ws);
  __shared__ unsigned short smA[PANEL];    // 64KB: first-stage weight panel
  __shared__ unsigned short smB[PANEL];    // 64KB: second-stage weight panel

  const unsigned ldsA = (unsigned)(uintptr_t)(&smA[0]);   // LDS byte offset
  const unsigned ldsB = (unsigned)(uintptr_t)(&smB[0]);

  const int tid  = threadIdx.x;
  const int wgid = blockIdx.x;                // 0 .. 23
  const int wv   = tid >> 5;                  // wave in WG: 0 .. 3
  const int lane = tid & 31;
  const int T    = *Tptr;
  unsigned bars  = 0;

  for (int t = 1; t < T; ++t) {
    // ---- Phase A: pre-activations. WG = (jt,nt); 4 waves = 4 M-tiles of 32 ----
    {
      const int jt = wgid >> 3;               // 0 = r, 1 = z, 2 = x@Wh
      const int nt = wgid & 7;                // 64-col group
      const int m0 = wv << 5;
      const int n0 = nt << 6;
      const unsigned short* W1 = (jt == 0) ? s.wp[1] : (jt == 1) ? s.wp[0] : s.wp[2];
      const unsigned short* W2 = (jt == 0) ? s.wp[4] : s.wp[3];   // Ur / Uz

      if (tid < 32) {                         // wave 0 drives the TDM
        tdm_load(ldsA, W1 + nt * PANEL, PANEL * 2);
        if (jt < 2) {
          tdm_load(ldsB, W2 + nt * PANEL, PANEL * 2);
          __builtin_amdgcn_s_wait_tensorcnt(1);     // bufA ready (in-order)
        } else {
          __builtin_amdgcn_s_wait_tensorcnt(0);
        }
      }
      __syncthreads();

      v8f c[2][4] = {};
      mm32x64_lds(s.xbf, smA, m0, lane, c);          // x @ W*
      if (jt < 2) {
        if (tid < 32) __builtin_amdgcn_s_wait_tensorcnt(0);
        __syncthreads();
        mm32x64_lds(s.hbf, smB, m0, lane, c);        // h @ U*
      }
#pragma unroll
      for (int mi = 0; mi < 2; ++mi)
#pragma unroll
        for (int sb = 0; sb < 4; ++sb)
#pragma unroll
          for (int j = 0; j < 8; ++j) {
            const int row = m0 + (mi << 4) + ((lane >> 4) << 3) + j;
            const int col = n0 + (sb << 4) + (lane & 15);
            const int idx = row * D_ + col;
            const float v = c[mi][sb][j];
            if (jt == 0) {
              float r = 1.0f / (1.0f + __expf(-(v + br[col])));
              s.rhbf[idx] = f2bf(r * s.hbuf[idx]);
            } else if (jt == 1) {
              s.zbuf[idx] = 1.0f / (1.0f + __expf(-(v + bz[col])));
            } else {
              s.hpart[idx] = v + bh[col];
            }
          }
    }
    bars++; grid_sync(s.cnt, bars * NBLK);

    // ---- Phase B: h_prop & h_new. Also pre-stage Wo for phase C (bufB) ----
    if (wgid < 16) {
      const int nt = wgid & 7;
      const int m0 = (((wgid >> 3) << 2) + wv) << 4;  // 16-row tile
      const int n0 = nt << 6;
      if (tid < 32) {
        tdm_load(ldsA, s.wp[5] + nt * PANEL, PANEL * 2);   // Uh
        tdm_load(ldsB, s.wp[6] + nt * PANEL, PANEL * 2);   // Wo (for phase C)
        __builtin_amdgcn_s_wait_tensorcnt(1);
      }
      __syncthreads();

      v8f c[4] = {};
      mm16x64_lds(s.rhbf, smA, m0, lane, c);
#pragma unroll
      for (int sb = 0; sb < 4; ++sb)
#pragma unroll
        for (int j = 0; j < 8; ++j) {
          const int row = m0 + ((lane >> 4) << 3) + j;
          const int col = n0 + (sb << 4) + (lane & 15);
          const int idx = row * D_ + col;
          float hp = tanhf(c[sb][j] + s.hpart[idx]);
          float z  = s.zbuf[idx];
          float h  = s.hbuf[idx];
          float hn = (1.0f - z) * h + z * hp;
          hn = fminf(5.0f, fmaxf(-5.0f, hn));
          s.hbuf[idx] = hn;
          s.hbf[idx]  = f2bf(hn);
        }
    }
    bars++; grid_sync(s.cnt, bars * NBLK);

    // ---- Phase C: out = h_new @ Wo + bo (Wo already resident in bufB) ----
    if (wgid < 16) {
      const int nt = wgid & 7;
      const int m0 = (((wgid >> 3) << 2) + wv) << 4;
      const int n0 = nt << 6;
      if (tid < 32) __builtin_amdgcn_s_wait_tensorcnt(0);
      __syncthreads();

      v8f c[4] = {};
      mm16x64_lds(s.hbf, smB, m0, lane, c);
#pragma unroll
      for (int sb = 0; sb < 4; ++sb)
#pragma unroll
        for (int j = 0; j < 8; ++j) {
          const int row = m0 + ((lane >> 4) << 3) + j;
          const int col = n0 + (sb << 4) + (lane & 15);
          float o = c[sb][j] + bo[col];
          out[(size_t)row * T * D_ + (size_t)t * D_ + col] = o;
          s.xbf[row * D_ + col] = f2bf(o);
        }
    }
    bars++; grid_sync(s.cnt, bars * NBLK);
  }
}

// ---------------- host launch ----------------
extern "C" void kernel_launch(void* const* d_in, const int* in_sizes, int n_in,
                              void* d_out, int out_size, void* d_ws, size_t ws_size,
                              hipStream_t stream) {
  const float* inputs = (const float*)d_in[0];
  const float* Wz = (const float*)d_in[1];
  const float* Wr = (const float*)d_in[2];
  const float* Wh = (const float*)d_in[3];
  const float* Uz = (const float*)d_in[4];
  const float* Ur = (const float*)d_in[5];
  const float* Uh = (const float*)d_in[6];
  const float* bz = (const float*)d_in[7];
  const float* br = (const float*)d_in[8];
  const float* bh = (const float*)d_in[9];
  const float* Wo = (const float*)d_in[10];
  const float* bo = (const float*)d_in[11];
  const int* Tptr = (const int*)d_in[12];
  float* out = (float*)d_out;

  pack_weights<<<7 * WSZ / 256, 256, 0, stream>>>(
      Wz, Wr, Wh, Uz, Ur, Uh, Wo, (unsigned short*)d_ws);
  init_state<<<(B_ * D_) / 256, 256, 0, stream>>>(inputs, out, Tptr, d_ws);
  gru_persistent<<<NBLK, TPB, 0, stream>>>(bz, br, bh, bo, out, Tptr, d_ws);
}
